// GAT_5050881540603
// MI455X (gfx1250) — matrix operand order repaired and checked by reading, not compiled, not run on hardware
//
#include <hip/hip_runtime.h>
#include <hip/hip_bf16.h>
#include <math.h>

// Problem constants (from reference)
#define GN     50000          // nodes
#define GE     800000         // edges (without self loops)
#define GET    (GE + GN)      // edges incl. self loops = 850000
#define GIN_C  128
#define GHID   64
#define GHEADS 4
#define GOUT_C 64
#define NCLS   40
#define NEG_SLOPE 0.2f

typedef __attribute__((ext_vector_type(16))) __bf16 v16bf;
typedef __attribute__((ext_vector_type(8)))  float  v8f;

static __device__ __forceinline__ unsigned fmap_ord(float f) {
  unsigned u = __float_as_uint(f);
  return (u & 0x80000000u) ? ~u : (u | 0x80000000u);
}
static __device__ __forceinline__ float funmap_ord(unsigned u) {
  unsigned b = (u & 0x80000000u) ? (u ^ 0x80000000u) : ~u;
  return __uint_as_float(b);
}
static __device__ __forceinline__ void edge_endpoints(const int* __restrict__ ei,
                                                      int e, int& s, int& d) {
  if (e < GE) { s = ei[e]; d = ei[GE + e]; }
  else        { s = e - GE; d = e - GE; }      // appended self loops
}

// ---------------------------------------------------------------------------
// Fill kernels (init for atomics; graph-capture safe, deterministic)
// ---------------------------------------------------------------------------
__global__ void fill_f32(float* p, float v, long long n) {
  long long i = (long long)blockIdx.x * blockDim.x + threadIdx.x;
  if (i < n) p[i] = v;
}
__global__ void fill_u32(unsigned* p, unsigned v, long long n) {
  long long i = (long long)blockIdx.x * blockDim.x + threadIdx.x;
  if (i < n) p[i] = v;
}

// ---------------------------------------------------------------------------
// Weight pre-pass: W[K][Nc] f32 (row-major) -> Wt[Nc][K] bf16 (transposed).
// Makes every WMMA B-fragment a single contiguous 32B load per lane.
// ---------------------------------------------------------------------------
__global__ void transpose_w_bf16(const float* __restrict__ W,
                                 __bf16* __restrict__ Wt, int K, int Nc) {
  int i = blockIdx.x * blockDim.x + threadIdx.x;
  if (i >= K * Nc) return;
  int k = i / Nc, n = i - k * Nc;
  Wt[(size_t)n * K + k] = (__bf16)W[i];
}

// ---------------------------------------------------------------------------
// WMMA GEMM: C[M,NC] = A[M,K] * W[K,NC]  (W pre-transposed to bf16 [NC][K]).
// One wave computes a 16x64 tile: one A fragment amortized over 4 WMMAs.
// Fully unrolled K loop (K, NC compile-time).
// VGPR layouts used here (A and B share one k order):
//   A (16x32 bf16): lanes 0-15 row r, K {k0+0..15}; lanes 16-31 row r,
//                   K {k0+16..31}
//   B (32x16 bf16): lanes 0-15 col r, K {k0+0..15}; lanes 16-31 col r, K {k0+16..31}
//   C (16x16 f32):  VGPR i -> row i + 8*half, col r
// ---------------------------------------------------------------------------
template <int K, int NC>
__global__ void gemm_wmma_bf16(const float* __restrict__ A,
                               const __bf16* __restrict__ Wt,
                               float* __restrict__ C, int M) {
  constexpr int NTN = NC / 64;
  const int wid  = blockIdx.x * (blockDim.x >> 5) + (threadIdx.x >> 5);
  const int lane = threadIdx.x & 31;
  const int ntiles = (M >> 4) * NTN;
  if (wid >= ntiles) return;               // uniform per-wave: EXEC stays all-1s

  const int tm   = wid / NTN;
  const int tn   = wid - tm * NTN;
  const int half = lane >> 4;              // 0: lanes 0-15, 1: lanes 16-31
  const int r    = lane & 15;

  const float*  arow = A + (size_t)(tm * 16 + r) * K;
  const __bf16* b0 = Wt + (size_t)(tn * 64 +  0 + r) * K;
  const __bf16* b1 = Wt + (size_t)(tn * 64 + 16 + r) * K;
  const __bf16* b2 = Wt + (size_t)(tn * 64 + 32 + r) * K;
  const __bf16* b3 = Wt + (size_t)(tn * 64 + 48 + r) * K;

  v8f c0 = {}, c1 = {}, c2 = {}, c3 = {};

#pragma unroll
  for (int k0 = 0; k0 < K; k0 += 32) {
    // A fragment (f32 -> bf16 in regs)
    const float4* alo = (const float4*)(arow + k0 + half * 16);
    const float4* ahi = (const float4*)(arow + k0 + half * 16 + 8);
    float4 a0 = alo[0], a1 = alo[1], a2 = ahi[0], a3 = ahi[1];
    float af[16] = {a0.x, a0.y, a0.z, a0.w, a1.x, a1.y, a1.z, a1.w,
                    a2.x, a2.y, a2.z, a2.w, a3.x, a3.y, a3.z, a3.w};
    v16bf a;
#pragma unroll
    for (int i = 0; i < 16; ++i) a[i] = (__bf16)af[i];

    // B fragments: 16 contiguous bf16 (32B, aligned) per lane per sub-tile
    const int kb = k0 + half * 16;
    v16bf vb0 = *(const v16bf*)(b0 + kb);
    v16bf vb1 = *(const v16bf*)(b1 + kb);
    v16bf vb2 = *(const v16bf*)(b2 + kb);
    v16bf vb3 = *(const v16bf*)(b3 + kb);

    c0 = __builtin_amdgcn_wmma_f32_16x16x32_bf16(false, a, false, vb0, (short)0, c0, false, false);
    c1 = __builtin_amdgcn_wmma_f32_16x16x32_bf16(false, a, false, vb1, (short)0, c1, false, false);
    c2 = __builtin_amdgcn_wmma_f32_16x16x32_bf16(false, a, false, vb2, (short)0, c2, false, false);
    c3 = __builtin_amdgcn_wmma_f32_16x16x32_bf16(false, a, false, vb3, (short)0, c3, false, false);
  }

  const int rowb = tm * 16 + half * 8;
  float* cp = C + (size_t)rowb * NC + tn * 64 + r;
#pragma unroll
  for (int i = 0; i < 8; ++i) {
    float* row = cp + (size_t)i * NC;
    row[0]  = c0[i];
    row[16] = c1[i];
    row[32] = c2[i];
    row[48] = c3[i];
  }
}

// ---------------------------------------------------------------------------
// alpha_s[n,h] = sum_c h[n,h,c]*a_src[h,c] ; alpha_d likewise
// ---------------------------------------------------------------------------
__global__ void alpha_kernel(const float* __restrict__ h,
                             const float* __restrict__ a_src,
                             const float* __restrict__ a_dst,
                             float* __restrict__ as, float* __restrict__ ad,
                             int H, int C) {
  long long i = (long long)blockIdx.x * blockDim.x + threadIdx.x;
  if (i >= (long long)GN * H) return;
  int node = (int)(i / H), head = (int)(i - (long long)node * H);
  const float* hp = h + (size_t)node * H * C + (size_t)head * C;
  const float* ws = a_src + (size_t)head * C;
  const float* wd = a_dst + (size_t)head * C;
  float s = 0.f, d = 0.f;
  for (int c = 0; c < C; ++c) { float v = hp[c]; s += v * ws[c]; d += v * wd[c]; }
  as[i] = s; ad[i] = d;
}

// ---------------------------------------------------------------------------
// Pass 1: segment-max of leaky_relu(alpha_s[src]+alpha_d[dst]) over dst
// ---------------------------------------------------------------------------
__global__ void edge_max_kernel(const int* __restrict__ ei,
                                const float* __restrict__ as,
                                const float* __restrict__ ad,
                                unsigned* __restrict__ emax, int H) {
  long long i = (long long)blockIdx.x * blockDim.x + threadIdx.x;
  if (i >= (long long)GET * H) return;
  int e = (int)(i / H), hd = (int)(i - (long long)e * H);
  int s, d; edge_endpoints(ei, e, s, d);
  float x = as[(size_t)s * H + hd] + ad[(size_t)d * H + hd];
  x = (x > 0.f) ? x : NEG_SLOPE * x;
  atomicMax(&emax[(size_t)d * H + hd], fmap_ord(x));
}

// ---------------------------------------------------------------------------
// Pass 2: ee = exp(e - emax[dst]); denom[dst] += ee; stash ee per edge
// ---------------------------------------------------------------------------
__global__ void edge_exp_kernel(const int* __restrict__ ei,
                                const float* __restrict__ as,
                                const float* __restrict__ ad,
                                const unsigned* __restrict__ emax,
                                float* __restrict__ ee,
                                float* __restrict__ denom, int H) {
  long long i = (long long)blockIdx.x * blockDim.x + threadIdx.x;
  if (i >= (long long)GET * H) return;
  int e = (int)(i / H), hd = (int)(i - (long long)e * H);
  int s, d; edge_endpoints(ei, e, s, d);
  float x = as[(size_t)s * H + hd] + ad[(size_t)d * H + hd];
  x = (x > 0.f) ? x : NEG_SLOPE * x;
  float m  = funmap_ord(emax[(size_t)d * H + hd]);
  float ev = expf(x - m);
  ee[i] = ev;
  atomicAdd(&denom[(size_t)d * H + hd], ev);
}

// ---------------------------------------------------------------------------
// Pass 3: agg[dst, h, c] += h[src, h, c] * ee/denom — one wave per edge.
// HC = H*C channels (C==64 both layers -> head = ch>>6).
// h (51MB) and agg (51MB) are L2-resident on MI455X's 192MB L2.
// ---------------------------------------------------------------------------
__global__ void edge_agg_kernel(const int* __restrict__ ei,
                                const float* __restrict__ h,
                                const float* __restrict__ ee,
                                const float* __restrict__ denom,
                                float* __restrict__ agg, int H, int HC) {
  int wid  = blockIdx.x * (blockDim.x >> 5) + (threadIdx.x >> 5);
  int lane = threadIdx.x & 31;
  if (wid >= GET) return;
  int s, d; edge_endpoints(ei, wid, s, d);
  const float* hs = h + (size_t)s * HC;
  float* ag = agg + (size_t)d * HC;
  for (int ch = lane; ch < HC; ch += 32) {
    int head = ch >> 6;  // C == 64
    float w = ee[(size_t)wid * H + head] /
              (denom[(size_t)d * H + head] + 1e-16f);
    atomicAdd(&ag[ch], hs[ch] * w);
  }
}

// out[i] = relu(agg[i] + bias[i % HC])   (in place OK)
__global__ void bias_relu_kernel(float* __restrict__ x,
                                 const float* __restrict__ b, int HC, long long n) {
  long long i = (long long)blockIdx.x * blockDim.x + threadIdx.x;
  if (i >= n) return;
  float v = x[i] + b[(int)(i % HC)];
  x[i] = v > 0.f ? v : 0.f;
}

// Global mean pool (partials via per-block LDS reduce + 64 atomics/block)
__global__ void pool_kernel(const float* __restrict__ h, float* __restrict__ pooled) {
  __shared__ float sm[256];
  int col = threadIdx.x & 63;
  int grp = threadIdx.x >> 6;
  float acc = 0.f;
  for (int node = blockIdx.x * 4 + grp; node < GN; node += gridDim.x * 4)
    acc += h[(size_t)node * 64 + col];
  sm[threadIdx.x] = acc;
  __syncthreads();
  if (grp == 0)
    atomicAdd(&pooled[col], sm[col] + sm[col + 64] + sm[col + 128] + sm[col + 192]);
}

// FC [64->40] + log_softmax, single block of 64 threads
__global__ void head_kernel(const float* __restrict__ pooled,
                            const float* __restrict__ fcw,
                            const float* __restrict__ fcb,
                            float* __restrict__ out) {
  __shared__ float pm[64];
  __shared__ float logits[NCLS];
  int t = threadIdx.x;
  pm[t] = pooled[t] * (1.0f / (float)GN);
  __syncthreads();
  if (t < NCLS) {
    float l = fcb[t];
#pragma unroll
    for (int c = 0; c < 64; ++c) l += pm[c] * fcw[c * NCLS + t];
    logits[t] = l;
  }
  __syncthreads();
  if (t == 0) {
    float mx = -INFINITY;
    for (int k = 0; k < NCLS; ++k) mx = fmaxf(mx, logits[k]);
    float se = 0.f;
    for (int k = 0; k < NCLS; ++k) se += expf(logits[k] - mx);
    float lse = logf(se) + mx;
    for (int k = 0; k < NCLS; ++k) out[k] = logits[k] - lse;
  }
}

// ---------------------------------------------------------------------------
extern "C" void kernel_launch(void* const* d_in, const int* in_sizes, int n_in,
                              void* d_out, int out_size, void* d_ws, size_t ws_size,
                              hipStream_t stream) {
  const float* x    = (const float*)d_in[0];
  const int*   ei   = (const int*)  d_in[1];
  const float* W1   = (const float*)d_in[2];
  const float* aS1  = (const float*)d_in[3];
  const float* aD1  = (const float*)d_in[4];
  const float* b1   = (const float*)d_in[5];
  const float* W2   = (const float*)d_in[6];
  const float* aS2  = (const float*)d_in[7];
  const float* aD2  = (const float*)d_in[8];
  const float* b2   = (const float*)d_in[9];
  const float* fcw  = (const float*)d_in[10];
  const float* fcb  = (const float*)d_in[11];
  float* out = (float*)d_out;

  // Workspace layout (floats). Peak ~119MB + 64KB bf16 weights.
  float*    ws    = (float*)d_ws;
  float*    h1    = ws;                                   // [N,256] (reused: h2 [N,64], agg2 [N,64])
  float*    agg1  = h1   + (size_t)GN * 256;              // [N,256] (becomes layer-2 input after relu)
  float*    as1   = agg1 + (size_t)GN * 256;              // [N,4]
  float*    ad1   = as1  + (size_t)GN * GHEADS;           // [N,4]
  float*    base  = ad1  + (size_t)GN * GHEADS;
  unsigned* emax  = (unsigned*)base;                      // [N,4] ordered-uint
  float*    denom = base + (size_t)GN * GHEADS;           // [N,4]
  float*    ee    = denom + (size_t)GN * GHEADS;          // [Etot,4]
  float*    pooled= ee    + (size_t)GET * GHEADS;         // [64]
  __bf16*   wt    = (__bf16*)(pooled + 64);               // [256,128] bf16 max (64KB), 32B-aligned
  float*    h2    = h1;                                   // [N,64]
  float*    agg2  = h1 + (size_t)GN * 64;                 // [N,64]

  const int T = 256;
  auto cdivll = [](long long a, long long b) { return (int)((a + b - 1) / b); };

  // ---------------- Layer 1 (H=4, C=64) ----------------
  // 0x007FFFFF == ordered-uint encoding of -inf (~bits(-inf))
  fill_u32<<<cdivll((long long)GN * GHEADS, T), T, 0, stream>>>(emax, 0x007FFFFFu, (long long)GN * GHEADS);
  fill_f32<<<cdivll((long long)GN * GHEADS, T), T, 0, stream>>>(denom, 0.f, (long long)GN * GHEADS);
  fill_f32<<<cdivll((long long)GN * 256, T), T, 0, stream>>>(agg1, 0.f, (long long)GN * 256);

  {  // h1 = x @ W1   (M=50000, K=128, Nc=256): 3125*4 = 12500 waves of 16x64 tiles
    transpose_w_bf16<<<cdivll(GIN_C * 256, T), T, 0, stream>>>(W1, wt, GIN_C, 256);
    int tiles = (GN / 16) * (256 / 64);
    gemm_wmma_bf16<GIN_C, 256><<<(tiles + 7) / 8, T, 0, stream>>>(x, wt, h1, GN);
  }
  alpha_kernel<<<cdivll((long long)GN * GHEADS, T), T, 0, stream>>>(h1, aS1, aD1, as1, ad1, GHEADS, GHID);
  edge_max_kernel<<<cdivll((long long)GET * GHEADS, T), T, 0, stream>>>(ei, as1, ad1, emax, GHEADS);
  edge_exp_kernel<<<cdivll((long long)GET * GHEADS, T), T, 0, stream>>>(ei, as1, ad1, emax, ee, denom, GHEADS);
  edge_agg_kernel<<<(GET + 7) / 8, T, 0, stream>>>(ei, h1, ee, denom, agg1, GHEADS, GHEADS * GHID);
  bias_relu_kernel<<<cdivll((long long)GN * 256, T), T, 0, stream>>>(agg1, b1, 256, (long long)GN * 256);

  // ---------------- Layer 2 (H=1, C=64) ----------------
  fill_u32<<<cdivll((long long)GN, T), T, 0, stream>>>(emax, 0x007FFFFFu, (long long)GN);
  fill_f32<<<cdivll((long long)GN, T), T, 0, stream>>>(denom, 0.f, (long long)GN);
  fill_f32<<<cdivll((long long)GN * 64, T), T, 0, stream>>>(agg2, 0.f, (long long)GN * 64);
  fill_f32<<<1, 64, 0, stream>>>(pooled, 0.f, 64);

  {  // h2 = h1o @ W2  (M=50000, K=256, Nc=64): 3125 waves of 16x64 tiles
    transpose_w_bf16<<<cdivll(256 * GOUT_C, T), T, 0, stream>>>(W2, wt, 256, GOUT_C);
    int tiles = (GN / 16) * (GOUT_C / 64);
    gemm_wmma_bf16<256, GOUT_C><<<(tiles + 7) / 8, T, 0, stream>>>(agg1, wt, h2, GN);
  }
  alpha_kernel<<<cdivll((long long)GN, T), T, 0, stream>>>(h2, aS2, aD2, as1, ad1, 1, GOUT_C);
  edge_max_kernel<<<cdivll((long long)GET, T), T, 0, stream>>>(ei, as1, ad1, emax, 1);
  edge_exp_kernel<<<cdivll((long long)GET, T), T, 0, stream>>>(ei, as1, ad1, emax, ee, denom, 1);
  edge_agg_kernel<<<(GET + 7) / 8, T, 0, stream>>>(ei, h2, ee, denom, agg2, 1, GOUT_C);
  bias_relu_kernel<<<cdivll((long long)GN * 64, T), T, 0, stream>>>(agg2, b2, 64, (long long)GN * 64);

  // ---------------- Pool + FC + log_softmax ----------------
  pool_kernel<<<128, T, 0, stream>>>(agg2, pooled);
  head_kernel<<<1, 64, 0, stream>>>(pooled, fcw, fcb, out);
  (void)in_sizes; (void)n_in; (void)out_size; (void)ws_size;
}
